// RestormerBlock_5153960755731
// MI455X (gfx1250) — compile-verified, matrix operand order checked by reading, not run on hardware
//
#include <hip/hip_runtime.h>
#include <math.h>

// ---------------- types ----------------
typedef __attribute__((ext_vector_type(16))) _Float16 v16h;
typedef __attribute__((ext_vector_type(8)))  _Float16 v8h;
typedef __attribute__((ext_vector_type(8)))  float    v8f;

union Frag { v16h v; v8h h[2]; };

// ---------------- problem constants ----------------
#define Cc     192
#define HEADSc 4
#define DHc    48          // C / HEADS
#define HIDc   510
#define HIDPc  512         // padded K for proj_out
#define Bc     2
#define Hc     192
#define Wc     192
#define HWc    36864
#define QKVc   576
#define QKVPc  608         // padded row stride for qkv (multiple of 32, room for h=3 v reads)
#define MP3c   1024        // padded M for proj_in

// ======================================================================
// instance-norm statistics: one block per (b,c) row of length HW
// ======================================================================
__global__ __launch_bounds__(256) void stats_kernel(const float* __restrict__ src,
                                                    float* __restrict__ mean,
                                                    float* __restrict__ rstd) {
  int row = blockIdx.x;
  const float* p = src + (size_t)row * HWc;
  float s = 0.f, ss = 0.f;
  for (int i = threadIdx.x; i < HWc; i += 256) {
    float v = p[i];
    s += v; ss += v * v;
  }
  __shared__ float sb[256], sb2[256];
  sb[threadIdx.x] = s; sb2[threadIdx.x] = ss;
  __syncthreads();
  for (int off = 128; off > 0; off >>= 1) {
    if (threadIdx.x < off) { sb[threadIdx.x] += sb[threadIdx.x + off]; sb2[threadIdx.x] += sb2[threadIdx.x + off]; }
    __syncthreads();
  }
  if (threadIdx.x == 0) {
    float m = sb[0] / (float)HWc;
    float v = sb2[0] / (float)HWc - m * m;
    mean[row] = m;
    rstd[row] = rsqrtf(fmaxf(v, 0.f) + 1e-5f);
  }
}

// ======================================================================
// normalize + pack to f16, transposing c-major [b][c][n] -> n-major [b][n][c]
// ======================================================================
__global__ __launch_bounds__(256) void pack_norm_kernel(const float* __restrict__ src,
                                                        const float* __restrict__ mean,
                                                        const float* __restrict__ rstd,
                                                        _Float16* __restrict__ dst) {
  size_t idx = (size_t)blockIdx.x * 256 + threadIdx.x;
  size_t total = (size_t)Bc * HWc * Cc;
  if (idx >= total) return;
  int c = (int)(idx % Cc);
  int n = (int)((idx / Cc) % HWc);
  int b = (int)(idx / ((size_t)Cc * HWc));
  int row = b * Cc + c;
  float v = (src[((size_t)row) * HWc + n] - mean[row]) * rstd[row];
  dst[idx] = (_Float16)v;     // idx == (b*HW + n)*C + c
}

// ======================================================================
// pack weights f32 [M,K] -> f16 [Mp,Kp] with zero pad
// ======================================================================
__global__ __launch_bounds__(256) void pack_w_kernel(const float* __restrict__ w,
                                                     _Float16* __restrict__ out,
                                                     int M, int K, int Mp, int Kp) {
  size_t idx = (size_t)blockIdx.x * 256 + threadIdx.x;
  size_t total = (size_t)Mp * Kp;
  if (idx >= total) return;
  int k = (int)(idx % Kp);
  int m = (int)(idx / Kp);
  out[idx] = (m < M && k < K) ? (_Float16)w[(size_t)m * K + k] : (_Float16)0.f;
}

// ======================================================================
// generic WMMA GEMM:  out[n][o] = sum_k act[b][n][k] * wt[o][k]  (+bias)
//   A-matrix = activation rows (M = spatial n), B-matrix = weight rows (N = o)
//   wave tile 64(M) x 32(N); block = 8 waves -> 256(M) x 64(N)
//   mode 0: f16 store n-major, stride oStride
//   mode 1: f32 store c-major d_out = resid + acc + bias
//   mode 2: f32 RMW  c-major d_out += acc + bias
// ======================================================================
__global__ __launch_bounds__(256) void wmma_gemm_kernel(const _Float16* __restrict__ act, int aStride,
                                                        const _Float16* __restrict__ wt, int K,
                                                        const float* __restrict__ bias, int Mreal,
                                                        int mode,
                                                        _Float16* __restrict__ out_h, int oStride,
                                                        float* __restrict__ out_f,
                                                        const float* __restrict__ resid) {
  int lane  = threadIdx.x & 31;
  int wid   = threadIdx.x >> 5;
  int l16   = lane & 15;
  int khalf = lane >> 4;
  int b     = blockIdx.z;
  int nbase = blockIdx.x * 256 + (wid & 3) * 64;
  int obase = blockIdx.y * 64 + (wid >> 2) * 32;
  const _Float16* actB = act + (size_t)b * HWc * aStride;

  v8f acc[4][2] = {};
  for (int kb = 0; kb < K; kb += 32) {
    Frag a[4], w2[2];
#pragma unroll
    for (int mt = 0; mt < 4; mt++) {
      const _Float16* p = actB + (size_t)(nbase + mt * 16 + l16) * aStride + kb + khalf * 8;
      a[mt].h[0] = *(const v8h*)p;
      a[mt].h[1] = *(const v8h*)(p + 16);
    }
#pragma unroll
    for (int ot = 0; ot < 2; ot++) {
      const _Float16* p = wt + (size_t)(obase + ot * 16 + l16) * K + kb + khalf * 8;
      w2[ot].h[0] = *(const v8h*)p;
      w2[ot].h[1] = *(const v8h*)(p + 16);
    }
#pragma unroll
    for (int mt = 0; mt < 4; mt++)
#pragma unroll
      for (int ot = 0; ot < 2; ot++)
        acc[mt][ot] = __builtin_amdgcn_wmma_f32_16x16x32_f16(
            false, a[mt].v, false, w2[ot].v, (short)0, acc[mt][ot], false, false);
  }

#pragma unroll
  for (int ot = 0; ot < 2; ot++) {
    int o = obase + ot * 16 + l16;
    float bv = (o < Mreal) ? bias[o] : 0.f;
#pragma unroll
    for (int mt = 0; mt < 4; mt++) {
#pragma unroll
      for (int r = 0; r < 8; r++) {
        int n = nbase + mt * 16 + r + khalf * 8;
        float val = acc[mt][ot][r] + bv;
        if (mode == 0) {
          out_h[((size_t)b * HWc + n) * oStride + o] = (_Float16)val;
        } else {
          size_t idx = ((size_t)b * Mreal + o) * HWc + n;
          if (mode == 1) out_f[idx] = resid[idx] + val;
          else           out_f[idx] += val;
        }
      }
    }
  }
}

// ======================================================================
// depthwise 3x3 over qkv (n-major in/out) + C-major q/k copy for score GEMM
// ======================================================================
__global__ __launch_bounds__(256) void dwconv_qkv_kernel(const _Float16* __restrict__ pre,   // [b][n][576]
                                                         const float* __restrict__ w,       // [576][9]
                                                         const float* __restrict__ bias,
                                                         _Float16* __restrict__ qkv_t,      // [b][n][608]
                                                         _Float16* __restrict__ qk_cm) {    // [b][384][HW]
  size_t idx = (size_t)blockIdx.x * 256 + threadIdx.x;
  size_t total = (size_t)Bc * HWc * QKVPc;
  if (idx >= total) return;
  int ch = (int)(idx % QKVPc);
  size_t t = idx / QKVPc;
  int n = (int)(t % HWc);
  int b = (int)(t / HWc);
  float acc = 0.f;
  if (ch < QKVc) {
    int y = n / Wc, x = n % Wc;
    acc = bias[ch];
#pragma unroll
    for (int ky = 0; ky < 3; ky++) {
      int yy = y + ky - 1;
      if (yy < 0 || yy >= Hc) continue;
#pragma unroll
      for (int kx = 0; kx < 3; kx++) {
        int xx = x + kx - 1;
        if (xx < 0 || xx >= Wc) continue;
        acc += w[ch * 9 + ky * 3 + kx] *
               (float)pre[((size_t)b * HWc + yy * Wc + xx) * QKVc + ch];
      }
    }
  }
  qkv_t[idx] = (_Float16)acc;
  if (ch < 2 * Cc) qk_cm[((size_t)b * 2 * Cc + ch) * HWc + n] = (_Float16)acc;
}

// ======================================================================
// per-row 1/L2-norm over spatial dim for q,k rows (B*384 rows)
// ======================================================================
__global__ __launch_bounds__(256) void rownorm_kernel(const _Float16* __restrict__ src,
                                                      float* __restrict__ rnorm) {
  int row = blockIdx.x;
  const _Float16* p = src + (size_t)row * HWc;
  float ss = 0.f;
  for (int i = threadIdx.x; i < HWc; i += 256) {
    float v = (float)p[i];
    ss += v * v;
  }
  __shared__ float sb[256];
  sb[threadIdx.x] = ss;
  __syncthreads();
  for (int off = 128; off > 0; off >>= 1) {
    if (threadIdx.x < off) sb[threadIdx.x] += sb[threadIdx.x + off];
    __syncthreads();
  }
  if (threadIdx.x == 0) rnorm[row] = 1.f / fmaxf(sqrtf(sb[0]), 1e-12f);
}

// ======================================================================
// score GEMM: raw q.k dot over spatial (K = HW) via WMMA, one wave per tile
// ======================================================================
__global__ __launch_bounds__(32) void score_gemm_kernel(const _Float16* __restrict__ qk,  // [b][384][HW]
                                                        float* __restrict__ scores) {     // [B*4*48][64]
  int lane = threadIdx.x;
  int l16 = lane & 15, khalf = lane >> 4;
  int t = blockIdx.x;
  int dt = t % 3;
  int ct = (t / 3) % 3;
  int bh = t / 9;                 // b*4 + h
  int h = bh % HEADSc, b = bh / HEADSc;
  const _Float16* qrow = qk + ((size_t)b * 2 * Cc + h * DHc + ct * 16 + l16) * HWc;
  const _Float16* krow = qk + ((size_t)b * 2 * Cc + Cc + h * DHc + dt * 16 + l16) * HWc;
  v8f acc = {};
  for (int kb = 0; kb < HWc; kb += 32) {
    Frag a, bb;
    a.h[0]  = *(const v8h*)(qrow + kb + khalf * 8);
    a.h[1]  = *(const v8h*)(qrow + kb + khalf * 8 + 16);
    bb.h[0] = *(const v8h*)(krow + kb + khalf * 8);
    bb.h[1] = *(const v8h*)(krow + kb + khalf * 8 + 16);
    acc = __builtin_amdgcn_wmma_f32_16x16x32_f16(false, a.v, false, bb.v, (short)0, acc, false, false);
  }
#pragma unroll
  for (int r = 0; r < 8; r++) {
    int c = ct * 16 + r + khalf * 8;
    int d = dt * 16 + l16;
    scores[((size_t)bh * DHc + c) * 64 + d] = acc[r];
  }
}

// ======================================================================
// fused scale (l2-norm factors + temperature) + softmax; pad 48->64 with 0
// one wave per attention row
// ======================================================================
__global__ __launch_bounds__(32) void softmax_kernel(const float* __restrict__ scores,
                                                     const float* __restrict__ rnorm,
                                                     const float* __restrict__ temp,
                                                     _Float16* __restrict__ attn_h) {
  int row = blockIdx.x;                 // B*4*48
  int c = row % DHc;
  int bh = row / DHc;
  int h = bh % HEADSc, b = bh / HEADSc;
  float rq = rnorm[b * 2 * Cc + h * DHc + c] * temp[h];
  int lane = threadIdx.x;
  int d0 = lane, d1 = lane + 32;
  float v0 = -3.0e38f, v1 = -3.0e38f;
  if (d0 < DHc) v0 = scores[(size_t)row * 64 + d0] * rq * rnorm[b * 2 * Cc + Cc + h * DHc + d0];
  if (d1 < DHc) v1 = scores[(size_t)row * 64 + d1] * rq * rnorm[b * 2 * Cc + Cc + h * DHc + d1];
  float mx = fmaxf(v0, v1);
  for (int i = 16; i > 0; i >>= 1) mx = fmaxf(mx, __shfl_xor(mx, i, 32));
  float e0 = (d0 < DHc) ? expf(v0 - mx) : 0.f;
  float e1 = (d1 < DHc) ? expf(v1 - mx) : 0.f;
  float s = e0 + e1;
  for (int i = 16; i > 0; i >>= 1) s += __shfl_xor(s, i, 32);
  float inv = 1.f / s;
  attn_h[(size_t)row * 64 + d0] = (_Float16)(e0 * inv);
  attn_h[(size_t)row * 64 + d1] = (_Float16)(e1 * inv);
}

// ======================================================================
// attn @ v : out_t[b][n][h*48+c] = sum_d v[n][d] * attn[c][d]  (K = 64, padded)
// one wave per 16x16 tile; 8 waves per block
// ======================================================================
__global__ __launch_bounds__(256) void attnv_kernel(const _Float16* __restrict__ qkv_t,  // [b][n][608]
                                                    const _Float16* __restrict__ attn_h, // [B*4*48][64]
                                                    _Float16* __restrict__ out_t) {      // [b][n][192]
  int lane = threadIdx.x & 31;
  int wid = threadIdx.x >> 5;
  int l16 = lane & 15, khalf = lane >> 4;
  int wg = blockIdx.x * 8 + wid;
  int ctile = wg % 3;
  int ntile = (wg / 3) % (HWc / 16);
  int h = (wg / (3 * (HWc / 16))) % HEADSc;
  int b = wg / (3 * (HWc / 16) * HEADSc);
  int nbase = ntile * 16, cbase = ctile * 16;

  const _Float16* arow = qkv_t + ((size_t)b * HWc + nbase + l16) * QKVPc + 2 * Cc + h * DHc;
  const _Float16* brow = attn_h + ((size_t)(b * HEADSc + h) * DHc + cbase + l16) * 64;

  v8f acc = {};
#pragma unroll
  for (int kb = 0; kb < 64; kb += 32) {
    Frag a, bb;
    a.h[0]  = *(const v8h*)(arow + kb + khalf * 8);
    a.h[1]  = *(const v8h*)(arow + kb + khalf * 8 + 16);
    bb.h[0] = *(const v8h*)(brow + kb + khalf * 8);
    bb.h[1] = *(const v8h*)(brow + kb + khalf * 8 + 16);
    acc = __builtin_amdgcn_wmma_f32_16x16x32_f16(false, a.v, false, bb.v, (short)0, acc, false, false);
  }
#pragma unroll
  for (int r = 0; r < 8; r++) {
    int n = nbase + r + khalf * 8;
    int c = cbase + l16;
    out_t[((size_t)b * HWc + n) * Cc + h * DHc + c] = (_Float16)acc[r];
  }
}

// ======================================================================
// fused depthwise 3x3 + exact GELU gate for GDFN; writes padded 512 channels
// ======================================================================
__global__ __launch_bounds__(256) void gate_kernel(const _Float16* __restrict__ pre,  // [b][n][1024]
                                                   const float* __restrict__ w,       // [1020][9]
                                                   const float* __restrict__ bias,
                                                   _Float16* __restrict__ gated) {    // [b][n][512]
  size_t idx = (size_t)blockIdx.x * 256 + threadIdx.x;
  size_t total = (size_t)Bc * HWc * HIDPc;
  if (idx >= total) return;
  int i = (int)(idx % HIDPc);
  size_t t = idx / HIDPc;
  int n = (int)(t % HWc);
  int b = (int)(t / HWc);
  float val = 0.f;
  if (i < HIDc) {
    int y = n / Wc, x = n % Wc;
    float a = bias[i];
    float g = bias[HIDc + i];
#pragma unroll
    for (int ky = 0; ky < 3; ky++) {
      int yy = y + ky - 1;
      if (yy < 0 || yy >= Hc) continue;
#pragma unroll
      for (int kx = 0; kx < 3; kx++) {
        int xx = x + kx - 1;
        if (xx < 0 || xx >= Wc) continue;
        size_t base = ((size_t)b * HWc + yy * Wc + xx) * MP3c;
        a += w[i * 9 + ky * 3 + kx] * (float)pre[base + i];
        g += w[(HIDc + i) * 9 + ky * 3 + kx] * (float)pre[base + HIDc + i];
      }
    }
    val = 0.5f * a * (1.f + erff(a * 0.70710678118f)) * g;
  }
  gated[idx] = (_Float16)val;
}

// ======================================================================
// host-side launch
// ======================================================================
extern "C" void kernel_launch(void* const* d_in, const int* in_sizes, int n_in,
                              void* d_out, int out_size, void* d_ws, size_t ws_size,
                              hipStream_t stream) {
  const float* x          = (const float*)d_in[0];
  const float* qkv_w      = (const float*)d_in[1];
  const float* qkv_b      = (const float*)d_in[2];
  const float* qkv_dw_w   = (const float*)d_in[3];
  const float* qkv_dw_b   = (const float*)d_in[4];
  const float* attn_out_w = (const float*)d_in[5];
  const float* attn_out_b = (const float*)d_in[6];
  const float* temperature= (const float*)d_in[7];
  const float* proj_in_w  = (const float*)d_in[8];
  const float* proj_in_b  = (const float*)d_in[9];
  const float* dw_w       = (const float*)d_in[10];
  const float* dw_b       = (const float*)d_in[11];
  const float* proj_out_w = (const float*)d_in[12];
  const float* proj_out_b = (const float*)d_in[13];
  float* out = (float*)d_out;
  char* ws = (char*)d_ws;

  size_t off = 0;
  auto take = [&](size_t bytes) { size_t o = off; off += (bytes + 255) & ~(size_t)255; return o; };

  float*    mean1   = (float*)(ws + take((size_t)Bc * Cc * 4));
  float*    rstd1   = (float*)(ws + take((size_t)Bc * Cc * 4));
  float*    mean2   = (float*)(ws + take((size_t)Bc * Cc * 4));
  float*    rstd2   = (float*)(ws + take((size_t)Bc * Cc * 4));
  float*    rnorm   = (float*)(ws + take((size_t)Bc * 2 * Cc * 4));
  float*    scores  = (float*)(ws + take((size_t)Bc * HEADSc * DHc * 64 * 4));
  _Float16* attn_h  = (_Float16*)(ws + take((size_t)Bc * HEADSc * DHc * 64 * 2));
  _Float16* w_qkv   = (_Float16*)(ws + take((size_t)QKVc * Cc * 2));
  _Float16* w_attn  = (_Float16*)(ws + take((size_t)Cc * Cc * 2));
  _Float16* w_pin   = (_Float16*)(ws + take((size_t)MP3c * Cc * 2));
  _Float16* w_pout  = (_Float16*)(ws + take((size_t)Cc * HIDPc * 2));
  _Float16* xn      = (_Float16*)(ws + take((size_t)Bc * HWc * Cc * 2));
  // region A: qkv_pre (85MB) + qkv_t (90MB); reused later by ffn_pre (151MB)
  char* regionA = ws + take((size_t)Bc * HWc * (QKVc + QKVPc) * 2);
  _Float16* qkv_pre = (_Float16*)regionA;
  _Float16* qkv_t   = (_Float16*)(regionA + (size_t)Bc * HWc * QKVc * 2);
  _Float16* ffn_pre = (_Float16*)regionA;
  // region B: qk_cm (57MB) + attnout (28MB); reused later by gated (76MB)
  char* regionB = ws + take((size_t)Bc * HWc * (2 * Cc + Cc) * 2);
  _Float16* qk_cm   = (_Float16*)regionB;
  _Float16* attnout = (_Float16*)(regionB + (size_t)Bc * HWc * 2 * Cc * 2);
  _Float16* gated   = (_Float16*)regionB;

  // ---- stage 1: norm + pack ----
  stats_kernel<<<Bc * Cc, 256, 0, stream>>>(x, mean1, rstd1);
  pack_norm_kernel<<<(Bc * HWc * Cc) / 256, 256, 0, stream>>>(x, mean1, rstd1, xn);

  // ---- weight packs (f16, padded) ----
  pack_w_kernel<<<((size_t)QKVc * Cc + 255) / 256, 256, 0, stream>>>(qkv_w, w_qkv, QKVc, Cc, QKVc, Cc);
  pack_w_kernel<<<((size_t)Cc * Cc + 255) / 256, 256, 0, stream>>>(attn_out_w, w_attn, Cc, Cc, Cc, Cc);
  pack_w_kernel<<<((size_t)MP3c * Cc + 255) / 256, 256, 0, stream>>>(proj_in_w, w_pin, 2 * HIDc, Cc, MP3c, Cc);
  pack_w_kernel<<<((size_t)Cc * HIDPc + 255) / 256, 256, 0, stream>>>(proj_out_w, w_pout, Cc, HIDc, Cc, HIDPc);

  // ---- qkv 1x1 GEMM (M=576, K=192) -> f16 n-major ----
  wmma_gemm_kernel<<<dim3(HWc / 256, QKVc / 64, Bc), 256, 0, stream>>>(
      xn, Cc, w_qkv, Cc, qkv_b, QKVc, 0, qkv_pre, QKVc, nullptr, nullptr);

  // ---- depthwise 3x3 on qkv (+ C-major q/k copy) ----
  dwconv_qkv_kernel<<<((size_t)Bc * HWc * QKVPc + 255) / 256, 256, 0, stream>>>(
      qkv_pre, qkv_dw_w, qkv_dw_b, qkv_t, qk_cm);

  // ---- channel attention ----
  rownorm_kernel<<<Bc * 2 * Cc, 256, 0, stream>>>(qk_cm, rnorm);
  score_gemm_kernel<<<Bc * HEADSc * 9, 32, 0, stream>>>(qk_cm, scores);
  softmax_kernel<<<Bc * HEADSc * DHc, 32, 0, stream>>>(scores, rnorm, temperature, attn_h);
  attnv_kernel<<<(Bc * HEADSc * (HWc / 16) * 3) / 8, 256, 0, stream>>>(qkv_t, attn_h, attnout);

  // ---- attn_out 1x1 + residual -> d_out (c-major fp32) ----
  wmma_gemm_kernel<<<dim3(HWc / 256, Cc / 64, Bc), 256, 0, stream>>>(
      attnout, Cc, w_attn, Cc, attn_out_b, Cc, 1, nullptr, 0, out, x);

  // ---- stage 2: GDFN ----
  stats_kernel<<<Bc * Cc, 256, 0, stream>>>(out, mean2, rstd2);
  pack_norm_kernel<<<(Bc * HWc * Cc) / 256, 256, 0, stream>>>(out, mean2, rstd2, xn);

  wmma_gemm_kernel<<<dim3(HWc / 256, MP3c / 64, Bc), 256, 0, stream>>>(
      xn, Cc, w_pin, Cc, proj_in_b, 2 * HIDc, 0, ffn_pre, MP3c, nullptr, nullptr);

  gate_kernel<<<((size_t)Bc * HWc * HIDPc) / 256, 256, 0, stream>>>(ffn_pre, dw_w, dw_b, gated);

  wmma_gemm_kernel<<<dim3(HWc / 256, Cc / 64, Bc), 256, 0, stream>>>(
      gated, HIDPc, w_pout, HIDPc, proj_out_b, Cc, 2, nullptr, 0, out, nullptr);
}
